// DeformableTransformer_57346403336774
// MI455X (gfx1250) — matrix operand order repaired and checked by reading, not compiled, not run on hardware
//
#include <hip/hip_runtime.h>
#include <math.h>

// ---------------- problem constants ----------------
#define B_    4
#define LQ    900
#define LIN   21760
#define D_    256
#define NH_   8
#define DH_   32
#define NL_   4
#define NP_   4
#define DFF_  1024
#define NROWS (B_*LQ)        // 3600 rows for query-side GEMMs

typedef __attribute__((ext_vector_type(16))) __bf16 v16bf;
typedef __attribute__((ext_vector_type(8)))  float  v8f;

union FragBF {
  v16bf v;
  unsigned short s[16];
  uint4 q[2];
};

__device__ __forceinline__ unsigned short f2bf(float f) {
  union { float f; unsigned u; } c; c.f = f;
  unsigned u = c.u;
  u += 0x7FFFu + ((u >> 16) & 1u);        // round-to-nearest-even
  return (unsigned short)(u >> 16);
}

// ---------------- plain f32 -> bf16 conversion (weights) ----------------
__global__ __launch_bounds__(256)
void cvt_bf16_kernel(const float* __restrict__ src, unsigned short* __restrict__ dst, int n) {
  int i = blockIdx.x * 256 + threadIdx.x;
  if (i < n) dst[i] = f2bf(src[i]);
}

// ---------------- gathered f32 (L,B,D) -> bf16 (B*L, D) conversion ----------------
// Each thread converts 8 contiguous channels of one row.
__global__ __launch_bounds__(256)
void cvt_gather_bf16_kernel(const float* __restrict__ src, unsigned short* __restrict__ dst,
                            int Lrows, int total8) {
  int i = blockIdx.x * 256 + threadIdx.x;
  if (i >= total8) return;
  const int d0 = (i & 31) * 8;                 // 32 chunks of 8 per 256-wide row
  const int r  = i >> 5;                       // logical row  b*Lrows + l
  const int b  = r / Lrows;
  const int l  = r - b * Lrows;
  const float* s = src + ((size_t)l * B_ + b) * D_ + d0;
  float4 a0 = *(const float4*)(s);
  float4 a1 = *(const float4*)(s + 4);
  unsigned short o[8] = { f2bf(a0.x), f2bf(a0.y), f2bf(a0.z), f2bf(a0.w),
                          f2bf(a1.x), f2bf(a1.y), f2bf(a1.z), f2bf(a1.w) };
  *(uint4*)(dst + (size_t)r * D_ + d0) = *(const uint4*)o;
}

// ---------------- WMMA GEMM:  C[M,N] = Abf @ Wbf^T + bias ----------------
// A, W are bf16 row-major (row stride K). Block = 8 waves; block tile = 128(M) x 32(N).
// The 32-column W strip is staged through LDS in 64-wide K chunks and shared by all waves.
// Each wave computes one 16x32 tile: one A fragment feeds two v_wmma_f32_16x16x32_bf16.
template <int K>
__global__ __launch_bounds__(256)
void gemm_wmma_bf16(const unsigned short* __restrict__ Abf,
                    const unsigned short* __restrict__ Wb,
                    const float* __restrict__ bias,
                    float* __restrict__ Cf, unsigned short* __restrict__ Cbf,
                    int Mtiles, int N, int relu)
{
  constexpr int LWS = 72;                       // padded LDS row stride (ushorts)
  __shared__ __align__(16) unsigned short lw[32 * LWS];

  const int tid   = threadIdx.x;
  const int wave  = tid >> 5;
  const int lane  = tid & 31;
  const int halfk = lane >> 4;                  // 0|1
  const int rc    = lane & 15;                  // A row / B col / C col within tile

  const int tileM_raw = blockIdx.y * 8 + wave;
  const bool active   = tileM_raw < Mtiles;
  const int tileM     = active ? tileM_raw : (Mtiles - 1);   // clamp: keep barriers uniform
  const int n0        = blockIdx.x * 32;

  const unsigned short* Arow = Abf + (size_t)(tileM * 16 + rc) * K;

  // staging assignment: thread i loads 8 ushorts of W row (n0 + i/8), k offset (i%8)*8
  const int sn = tid >> 3;
  const int sk = (tid & 7) * 8;
  const unsigned short* Wsrc = Wb + (size_t)(n0 + sn) * K + sk;

  const float bv0 = bias[n0 + rc];
  const float bv1 = bias[n0 + 16 + rc];
  v8f acc0, acc1;
  #pragma unroll
  for (int j = 0; j < 8; ++j) { acc0[j] = bv0; acc1[j] = bv1; }

  const int kk0 = halfk * 8;
  for (int k0 = 0; k0 < K; k0 += 64) {
    // cooperative stage of W[n0..n0+31][k0..k0+63] into LDS
    uint4 wchunk = *(const uint4*)(Wsrc + k0);
    if (k0 + 64 < K) {
      __builtin_prefetch((const void*)(Arow + k0 + 64), 0, 1);   // global_prefetch_b8
      __builtin_prefetch((const void*)(Wsrc + k0 + 64), 0, 1);
    }
    __syncthreads();                              // prior chunk fully consumed
    *(uint4*)(lw + sn * LWS + sk) = wchunk;
    __syncthreads();                              // chunk visible to all waves

    #pragma unroll
    for (int kk = 0; kk < 64; kk += 32) {
      // A fragment (16x32): elems 0-7 -> K = kk+halfk*8.., elems 8-15 -> K = kk+16+halfk*8..
      FragBF fa;
      fa.q[0] = *(const uint4*)(Arow + k0 + kk + kk0);
      fa.q[1] = *(const uint4*)(Arow + k0 + kk + 16 + kk0);
      // B fragments from LDS: lane holds col rc (+16), elems 0-7 -> K = kk+halfk*8.., elems 8-15 -> K = kk+16+halfk*8..
      FragBF fb0, fb1;
      const unsigned short* bp0 = lw + rc * LWS + kk + kk0;
      const unsigned short* bp1 = lw + (16 + rc) * LWS + kk + kk0;
      fb0.q[0] = *(const uint4*)(bp0);
      fb0.q[1] = *(const uint4*)(bp0 + 16);
      fb1.q[0] = *(const uint4*)(bp1);
      fb1.q[1] = *(const uint4*)(bp1 + 16);

      acc0 = __builtin_amdgcn_wmma_f32_16x16x32_bf16(false, fa.v, false, fb0.v,
                                                     (short)0, acc0, false, false);
      acc1 = __builtin_amdgcn_wmma_f32_16x16x32_bf16(false, fa.v, false, fb1.v,
                                                     (short)0, acc1, false, false);
    }
  }

  if (active) {
    const int m_b = tileM * 16 + halfk * 8;       // element j -> row m_b + j
    #pragma unroll
    for (int j = 0; j < 8; ++j) {
      float v0 = acc0[j], v1 = acc1[j];
      if (relu) { v0 = v0 > 0.0f ? v0 : 0.0f; v1 = v1 > 0.0f ? v1 : 0.0f; }
      const size_t row = (size_t)(m_b + j) * N;
      if (Cf)  { Cf[row + n0 + rc] = v0;        Cf[row + n0 + 16 + rc] = v1; }
      if (Cbf) { Cbf[row + n0 + rc] = f2bf(v0); Cbf[row + n0 + 16 + rc] = f2bf(v1); }
    }
  }
}

// ---------------- softmax over the 16 (level,point) logits per (b,l,h), in place ----------------
__global__ __launch_bounds__(256)
void softmax16_kernel(float* __restrict__ attn) {
  int t = blockIdx.x * 256 + threadIdx.x;          // (b*Lq+l)*NH + h
  if (t >= NROWS * NH_) return;
  float* row = attn + (size_t)t * 16;
  float mx = row[0];
  #pragma unroll
  for (int i = 1; i < 16; ++i) mx = fmaxf(mx, row[i]);
  float s = 0.f, e[16];
  #pragma unroll
  for (int i = 0; i < 16; ++i) { e[i] = __expf(row[i] - mx); s += e[i]; }
  float inv = 1.0f / s;
  #pragma unroll
  for (int i = 0; i < 16; ++i) row[i] = e[i] * inv;
}

// ---------------- MS-deformable bilinear sampling ----------------
// value: (B, Lin, D) row-major f32; one wave per (b,l,h), lane = channel d in [0,32)
// Writes f32 result and a bf16 copy (A-operand of the following output projection).
__global__ __launch_bounds__(256)
void msda_sample_kernel(const float* __restrict__ value, const float* __restrict__ off,
                        const float* __restrict__ attn, const float* __restrict__ refp,
                        float* __restrict__ out, unsigned short* __restrict__ outbf) {
  const int t = blockIdx.x * 8 + (threadIdx.x >> 5);     // (b*Lq+l)*NH + h
  const int lane = threadIdx.x & 31;
  if (t >= NROWS * NH_) return;
  const int h  = t & 7;
  const int bl = t >> 3;
  const int l  = bl % LQ;
  const int b  = bl / LQ;

  const int HWs[NL_][3] = {{128,128,0},{64,64,16384},{32,32,20480},{16,16,21504}};
  const float* offr = off  + (size_t)bl * (NH_*NL_*NP_*2);
  const float* attr = attn + (size_t)bl * (NH_*NL_*NP_);
  const size_t vbase = ((size_t)b * LIN) * D_ + h * DH_ + lane;

  float acc = 0.0f;
  #pragma unroll
  for (int lv = 0; lv < NL_; ++lv) {
    const int H = HWs[lv][0], W = HWs[lv][1], st = HWs[lv][2];
    const float rx = refp[(((size_t)l * B_ + b) * NL_ + lv) * 2 + 0];
    const float ry = refp[(((size_t)l * B_ + b) * NL_ + lv) * 2 + 1];
    #pragma unroll
    for (int p = 0; p < NP_; ++p) {
      const int oc = ((h * NL_ + lv) * NP_ + p) * 2;
      const float aw = attr[(h * NL_ + lv) * NP_ + p];
      const float x = rx * (float)W + offr[oc]     - 0.5f;
      const float y = ry * (float)H + offr[oc + 1] - 0.5f;
      const float xf = floorf(x), yf = floorf(y);
      const int x0 = (int)xf, y0 = (int)yf;
      const float wx1 = x - xf, wy1 = y - yf;
      const float wx0 = 1.0f - wx1, wy0 = 1.0f - wy1;
      float s = 0.0f;
      #pragma unroll
      for (int ty = 0; ty < 2; ++ty) {
        #pragma unroll
        for (int tx = 0; tx < 2; ++tx) {
          const int xi = x0 + tx, yi = y0 + ty;
          const float w = (tx ? wx1 : wx0) * (ty ? wy1 : wy0);
          if (xi >= 0 && xi < W && yi >= 0 && yi < H) {
            s += w * value[vbase + (size_t)(st + yi * W + xi) * D_];
          }
        }
      }
      acc += aw * s;
    }
  }
  const size_t oidx = (size_t)bl * D_ + h * DH_ + lane;
  out[oidx]   = acc;
  outbf[oidx] = f2bf(acc);
}

// ---------------- residual add + LayerNorm (one wave per row of 256) ----------------
// xmode==1: X is (Lq,B,D) layout (tgt); omode==1: write (Lq,B,D) layout (final output)
// outbf (optional): bf16 copy for the next GEMM's A operand.
__global__ __launch_bounds__(256)
void add_ln_kernel(const float* __restrict__ X, int xmode, const float* __restrict__ Y,
                   const float* __restrict__ g, const float* __restrict__ bta,
                   float* __restrict__ outp, unsigned short* __restrict__ outbf, int omode) {
  const int wave = threadIdx.x >> 5, lane = threadIdx.x & 31;
  const int r = blockIdx.x * 8 + wave;
  if (r >= NROWS) return;
  const int b = r / LQ, l = r % LQ;
  const float* xr = (xmode == 1) ? (X + (size_t)(l * B_ + b) * D_) : (X + (size_t)r * D_);
  const float* yr = Y + (size_t)r * D_;
  const int d0 = lane * 8;

  float v[8]; float s = 0.0f;
  #pragma unroll
  for (int i = 0; i < 8; ++i) { v[i] = xr[d0 + i] + yr[d0 + i]; s += v[i]; }
  #pragma unroll
  for (int m = 16; m >= 1; m >>= 1) s += __shfl_xor(s, m, 32);
  const float mu = s * (1.0f / D_);
  float s2 = 0.0f;
  #pragma unroll
  for (int i = 0; i < 8; ++i) { float d = v[i] - mu; s2 += d * d; }
  #pragma unroll
  for (int m = 16; m >= 1; m >>= 1) s2 += __shfl_xor(s2, m, 32);
  const float rs = rsqrtf(s2 * (1.0f / D_) + 1e-5f);

  float* orow = (omode == 1) ? (outp + (size_t)(l * B_ + b) * D_) : (outp + (size_t)r * D_);
  #pragma unroll
  for (int i = 0; i < 8; ++i) {
    const float o = (v[i] - mu) * rs * g[d0 + i] + bta[d0 + i];
    orow[d0 + i] = o;
    if (outbf) outbf[(size_t)r * D_ + d0 + i] = f2bf(o);
  }
}

// ---------------- host orchestration ----------------
extern "C" void kernel_launch(void* const* d_in, const int* in_sizes, int n_in,
                              void* d_out, int out_size, void* d_ws, size_t ws_size,
                              hipStream_t stream) {
  const float* tgt    = (const float*)d_in[0];    // (Lq,B,D)
  const float* refp   = (const float*)d_in[1];    // (Lq,B,NL,2)
  const float* mem    = (const float*)d_in[2];    // (Lin,B,D)
  const float* W_off  = (const float*)d_in[3];
  const float* b_off  = (const float*)d_in[4];
  const float* W_attn = (const float*)d_in[5];
  const float* b_attn = (const float*)d_in[6];
  const float* W_val  = (const float*)d_in[7];
  const float* b_val  = (const float*)d_in[8];
  const float* W_out  = (const float*)d_in[9];
  const float* b_out  = (const float*)d_in[10];
  const float* ln1_g  = (const float*)d_in[11];
  const float* ln1_b  = (const float*)d_in[12];
  const float* W1     = (const float*)d_in[13];
  const float* b1     = (const float*)d_in[14];
  const float* W2     = (const float*)d_in[15];
  const float* b2     = (const float*)d_in[16];
  const float* ln3_g  = (const float*)d_in[17];
  const float* ln3_b  = (const float*)d_in[18];
  float* out = (float*)d_out;                     // (Lq,B,D)

  // ---- workspace carve-up ----
  char* base = (char*)d_ws;
  size_t woff = 0;
  auto carve = [&](size_t bytes) -> char* {
    char* p = base + woff;
    woff = (woff + bytes + 255) & ~(size_t)255;
    return p;
  };
  unsigned short* wval_bf  = (unsigned short*)carve((size_t)D_ * D_ * 2);
  unsigned short* woff_bf  = (unsigned short*)carve((size_t)D_ * D_ * 2);
  unsigned short* wattn_bf = (unsigned short*)carve((size_t)(NH_*NL_*NP_) * D_ * 2);
  unsigned short* wout_bf  = (unsigned short*)carve((size_t)D_ * D_ * 2);
  unsigned short* w1_bf    = (unsigned short*)carve((size_t)DFF_ * D_ * 2);
  unsigned short* w2_bf    = (unsigned short*)carve((size_t)D_ * DFF_ * 2);
  unsigned short* membf    = (unsigned short*)carve((size_t)B_ * LIN * D_ * 2);  // (B*Lin, D)
  unsigned short* qbf      = (unsigned short*)carve((size_t)NROWS * D_ * 2);
  unsigned short* msdabf   = (unsigned short*)carve((size_t)NROWS * D_ * 2);
  unsigned short* tbf      = (unsigned short*)carve((size_t)NROWS * D_ * 2);
  unsigned short* hidbf    = (unsigned short*)carve((size_t)NROWS * DFF_ * 2);
  float* valbuf  = (float*)carve((size_t)B_ * LIN * D_ * 4);   // (B,Lin,D)
  float* offbuf  = (float*)carve((size_t)NROWS * D_ * 4);
  float* attnbuf = (float*)carve((size_t)NROWS * 128 * 4);
  float* msdabuf = (float*)carve((size_t)NROWS * D_ * 4);
  float* cabuf   = (float*)carve((size_t)NROWS * D_ * 4);
  float* tbuf    = (float*)carve((size_t)NROWS * D_ * 4);
  float* t2buf   = (float*)carve((size_t)NROWS * D_ * 4);
  (void)ws_size; (void)in_sizes; (void)n_in; (void)out_size;

  // ---- weights -> bf16 ----
  auto cvt = [&](const float* src, unsigned short* dst, int n) {
    cvt_bf16_kernel<<<(n + 255) / 256, 256, 0, stream>>>(src, dst, n);
  };
  cvt(W_val,  wval_bf,  D_ * D_);
  cvt(W_off,  woff_bf,  D_ * D_);
  cvt(W_attn, wattn_bf, 128 * D_);
  cvt(W_out,  wout_bf,  D_ * D_);
  cvt(W1,     w1_bf,    DFF_ * D_);
  cvt(W2,     w2_bf,    D_ * DFF_);

  // ---- activations -> bf16 (batch-major gather of (L,B,D) tensors) ----
  {
    int t8m = (B_ * LIN) * (D_ / 8);
    cvt_gather_bf16_kernel<<<(t8m + 255) / 256, 256, 0, stream>>>(mem, membf, LIN, t8m);
    int t8q = NROWS * (D_ / 8);
    cvt_gather_bf16_kernel<<<(t8q + 255) / 256, 256, 0, stream>>>(tgt, qbf, LQ, t8q);
  }

  // ---- value projection: (B*Lin,256) = membf @ Wval^T + bval ----
  {
    int Mtiles = (B_ * LIN) / 16;                      // 5440
    dim3 grid(D_ / 32, (Mtiles + 7) / 8);
    gemm_wmma_bf16<D_><<<grid, 256, 0, stream>>>(membf, wval_bf, b_val,
                                                 valbuf, nullptr, Mtiles, D_, 0);
  }
  // ---- sampling offsets / attention logits ----
  {
    int Mtiles = NROWS / 16;                           // 225
    dim3 g1(D_ / 32, (Mtiles + 7) / 8);
    gemm_wmma_bf16<D_><<<g1, 256, 0, stream>>>(qbf, woff_bf, b_off,
                                               offbuf, nullptr, Mtiles, D_, 0);
    dim3 g2(128 / 32, (Mtiles + 7) / 8);
    gemm_wmma_bf16<D_><<<g2, 256, 0, stream>>>(qbf, wattn_bf, b_attn,
                                               attnbuf, nullptr, Mtiles, 128, 0);
  }
  softmax16_kernel<<<(NROWS * NH_ + 255) / 256, 256, 0, stream>>>(attnbuf);

  // ---- deformable bilinear sampling (f32 + bf16 outputs) ----
  msda_sample_kernel<<<(NROWS * NH_) / 8, 256, 0, stream>>>(valbuf, offbuf, attnbuf,
                                                            refp, msdabuf, msdabf);
  // ---- output projection ----
  {
    int Mtiles = NROWS / 16;
    dim3 grid(D_ / 32, (Mtiles + 7) / 8);
    gemm_wmma_bf16<D_><<<grid, 256, 0, stream>>>(msdabf, wout_bf, b_out,
                                                 cabuf, nullptr, Mtiles, D_, 0);
  }
  // ---- LN1: t = LN(tgt + ca), also bf16 copy for FFN ----
  add_ln_kernel<<<(NROWS + 7) / 8, 256, 0, stream>>>(tgt, 1, cabuf, ln1_g, ln1_b,
                                                     tbuf, tbf, 0);
  // ---- FFN ----
  {
    int Mtiles = NROWS / 16;
    dim3 g1(DFF_ / 32, (Mtiles + 7) / 8);
    gemm_wmma_bf16<D_><<<g1, 256, 0, stream>>>(tbf, w1_bf, b1,
                                               nullptr, hidbf, Mtiles, DFF_, 1);
    dim3 g2(D_ / 32, (Mtiles + 7) / 8);
    gemm_wmma_bf16<DFF_><<<g2, 256, 0, stream>>>(hidbf, w2_bf, b2,
                                                 t2buf, nullptr, Mtiles, D_, 0);
  }
  // ---- LN3 -> output in (Lq,B,D) ----
  add_ln_kernel<<<(NROWS + 7) / 8, 256, 0, stream>>>(tbuf, 0, t2buf, ln3_g, ln3_b,
                                                     out, nullptr, 1);
}
